// Net_88691074662770
// MI455X (gfx1250) — matrix operand order repaired and checked by reading, not compiled, not run on hardware
//
#include <hip/hip_runtime.h>

// ---------------------------------------------------------------------------
// Problem constants (from the reference):
//   B=32768 rows, NUM=40 in/out features, HID=60 trunk dim, 80 heads.
//   out[b,h,d] = relu( sum_k relu(x@W1^T+b1)[b,k] * Wh[h,k,d] + bh[h,d] )
// Strategy: f16 WMMA (16x16x32, f32 accumulate), K padded 60->64 (2 k-blocks),
// N=40 padded to 3 tiles of 16. Memory-bound on the 419 MB f32 output stream;
// B fragments register-blocked across 4 row tiles to keep L2 weight traffic
// (~0.25 GB) well under the HBM store floor (~18 us @ 23.3 TB/s).
// ---------------------------------------------------------------------------

typedef __attribute__((ext_vector_type(16))) _Float16 v16h;
typedef __attribute__((ext_vector_type(8)))  float    v8f;

#define B_ROWS   32768
#define NUMF     40
#define HID      60
#define NHEADS   80
#define OUTCOLS  (NHEADS * NUMF)      // 3200
#define ROWTILES (B_ROWS / 16)        // 2048
#define RTPW     4                    // row tiles per wave (register blocking)

// ---------------------------------------------------------------------------
// Kernel 1: trunk  h = relu(x @ W1^T + b1), emitted as f16 in WMMA A-fragment
// swizzled layout: hswz[rowTile][kb][lane][e], 16 halves (32 B) per lane.
// A-fragment mapping (ISA 16-bit A 16x32 table):
//   M = lane & 15
//   K = (e&7) + 8*(lane>=16) + 16*(e>=8) + 32*kb      (K in [0,64), pad >=60 with 0)
// ---------------------------------------------------------------------------
__global__ void __launch_bounds__(256)
trunk_swz_kernel(const float* __restrict__ x,
                 const float* __restrict__ W1,
                 const float* __restrict__ b1,
                 _Float16* __restrict__ hswz) {
    const int tid     = blockIdx.x * 256 + threadIdx.x;   // 0 .. 2048*64-1
    const int lane    = tid & 31;
    const int kb      = (tid >> 5) & 1;
    const int rowTile = tid >> 6;
    const int row     = rowTile * 16 + (lane & 15);

    // Cache the 40-element x row in registers (reused for up to 16 dot products).
    float xv[NUMF];
    const float* __restrict__ xr = x + (size_t)row * NUMF;
#pragma unroll
    for (int j = 0; j < NUMF; ++j) xv[j] = xr[j];

    v16h frag;
#pragma unroll
    for (int e = 0; e < 16; ++e) {
        const int k = (e & 7) + ((lane >> 4) << 3) + ((e >> 3) << 4) + (kb << 5);
        float acc = 0.0f;
        if (k < HID) {
            acc = b1[k];
            const float* __restrict__ w = W1 + k * NUMF;   // 9.6 KB total, L1-resident
#pragma unroll
            for (int j = 0; j < NUMF; ++j) acc = fmaf(xv[j], w[j], acc);
            acc = acc > 0.0f ? acc : 0.0f;                 // ReLU
        }
        frag[e] = (_Float16)acc;
    }
    *(v16h*)(hswz + (size_t)tid * 16) = frag;
}

// ---------------------------------------------------------------------------
// Kernel 2: convert Wh[80][60][40] (f32) into f16 WMMA B-fragments:
//   wswz[head][dtile(3)][kb(2)][lane][e]
// B-fragment mapping (K x N, lane = column):
//   N = dtile*16 + (lane & 15)                (pad N>=40 with 0)
//   K = (e&7) + 8*(lane>=16) + 16*(e>=8) + 32*kb   (pad K>=60 with 0)
// Total 0.5 MB -> L2-resident, broadcast to every wave of the GEMM kernel.
// ---------------------------------------------------------------------------
__global__ void __launch_bounds__(256)
wh_swz_kernel(const float* __restrict__ Wh,
              _Float16* __restrict__ wswz) {
    const int tid  = blockIdx.x * 256 + threadIdx.x;      // 0 .. 15359
    const int lane = tid & 31;
    const int kb   = (tid >> 5) & 1;
    const int t    = (tid >> 6) % 3;
    const int hd   = tid / 192;
    const int d    = t * 16 + (lane & 15);

    v16h frag;
#pragma unroll
    for (int e = 0; e < 16; ++e) {
        const int k = (e & 7) + ((lane >> 4) << 3) + ((e >> 3) << 4) + (kb << 5);
        float v = 0.0f;
        if (d < NUMF && k < HID) v = Wh[((size_t)hd * HID + k) * NUMF + d];
        frag[e] = (_Float16)v;
    }
    *(v16h*)(wswz + (size_t)tid * 16) = frag;             // layout index == tid
}

// ---------------------------------------------------------------------------
// Kernel 3: batched-heads GEMM via v_wmma_f32_16x16x32_f16.
// Grid: (ROWTILES/(8*RTPW), 10). Block: 256 threads = 8 waves. Wave w owns
// row tiles [4*(blockIdx.x*8+w) .. +3]; blockIdx.y picks a group of 8 heads.
// Loop order: (head, d-tile) outer with the B fragment pair held in registers,
// row tile inner -> each B fragment is fetched once per wave and reused 4x,
// cutting L2 weight traffic 4x. A fragments (8 x v16h) stay in registers for
// the whole kernel. Epilogue fuses bias + ReLU and streams f32 results with
// non-temporal stores (output >> 192 MB L2).
// C/D layout: row = rowBase + r + 8*(lane>=16), col = lane & 15.
// ---------------------------------------------------------------------------
__global__ void __launch_bounds__(256)
heads_wmma_kernel(const _Float16* __restrict__ hswz,
                  const _Float16* __restrict__ wswz,
                  const float* __restrict__ bh,
                  float* __restrict__ out) {
    const int lane     = threadIdx.x & 31;
    const int wave     = threadIdx.x >> 5;
    const int rtBase   = (blockIdx.x * 8 + wave) * RTPW;   // first of 4 row tiles
    const int headBase = blockIdx.y * 8;

    // Preload A fragments for 4 row tiles x 2 k-blocks (64 VGPRs).
    v16h a[RTPW][2];
#pragma unroll
    for (int r = 0; r < RTPW; ++r) {
#pragma unroll
        for (int kb = 0; kb < 2; ++kb) {
            a[r][kb] = *(const v16h*)(hswz +
                ((size_t)((rtBase + r) * 2 + kb) * 32 + lane) * 16);
        }
    }

    const int rOff = (lane >> 4) << 3;   // +0 or +8 row offset within a C tile
    const int nCol = lane & 15;          // output column within the d-tile

    for (int hh = 0; hh < 8; ++hh) {
        const int hd = headBase + hh;
#pragma unroll
        for (int t = 0; t < 3; ++t) {
            const _Float16* __restrict__ bp =
                wswz + ((size_t)((hd * 3 + t) * 2) * 32 + lane) * 16;
            const v16h bf0 = *(const v16h*)(bp);            // K =  0..31
            const v16h bf1 = *(const v16h*)(bp + 32 * 16);  // K = 32..63

            const int  d     = t * 16 + nCol;
            const bool valid = (d < NUMF);
            const float bias = valid ? bh[hd * NUMF + d] : 0.0f;

#pragma unroll
            for (int r = 0; r < RTPW; ++r) {
                v8f acc = {};
                // 8 args: (neg_a, A, neg_b, B, c_mod, C, reuse_a, reuse_b)
                acc = __builtin_amdgcn_wmma_f32_16x16x32_f16(
                          false, a[r][0], false, bf0, (short)0, acc, false, false);
                acc = __builtin_amdgcn_wmma_f32_16x16x32_f16(
                          false, a[r][1], false, bf1, (short)0, acc, false, false);

                if (valid) {                                  // skip N-padding cols
                    float* __restrict__ op = out +
                        (size_t)((rtBase + r) * 16 + rOff) * OUTCOLS + hd * NUMF + d;
#pragma unroll
                    for (int e = 0; e < 8; ++e) {
                        float v = acc[e] + bias;
                        v = v > 0.0f ? v : 0.0f;              // fused ReLU
                        __builtin_nontemporal_store(v, op + (size_t)e * OUTCOLS);
                    }
                }
            }
        }
    }
}

// ---------------------------------------------------------------------------
// Launch: three kernels on one stream (in-order). Scratch partition:
//   hswz: 2048*64*16 halves = 4 MB   at d_ws + 0
//   wswz: 15360*16   halves = 0.5 MB at d_ws + 4 MB
// ---------------------------------------------------------------------------
extern "C" void kernel_launch(void* const* d_in, const int* in_sizes, int n_in,
                              void* d_out, int out_size, void* d_ws, size_t ws_size,
                              hipStream_t stream) {
    const float* x  = (const float*)d_in[0];   // [32768,40]
    const float* W1 = (const float*)d_in[1];   // [60,40]
    const float* b1 = (const float*)d_in[2];   // [60]
    const float* Wh = (const float*)d_in[3];   // [80,60,40]
    const float* bh = (const float*)d_in[4];   // [80,40]
    float* out = (float*)d_out;                // [32768,80,40]

    _Float16* hswz = (_Float16*)d_ws;
    _Float16* wswz = (_Float16*)((char*)d_ws + (size_t)ROWTILES * 64 * 16 * 2);

    trunk_swz_kernel<<<(ROWTILES * 64) / 256, 256, 0, stream>>>(x, W1, b1, hswz);
    wh_swz_kernel<<<(NHEADS * 3 * 2 * 32) / 256, 256, 0, stream>>>(Wh, wswz);
    heads_wmma_kernel<<<dim3(ROWTILES / (8 * RTPW), 10), 256, 0, stream>>>(
        hswz, wswz, bh, out);
}
